// StructedGcnConv_51110110822772
// MI455X (gfx1250) — matrix operand order repaired and checked by reading, not compiled, not run on hardware
//
#include <hip/hip_runtime.h>
#include <hip/hip_fp16.h>

// ---------------------------------------------------------------------------
// StructedGcnConv for MI455X (gfx1250): all linear layers run on
// v_wmma_f32_16x16x32_f16; f16 pair-encoding tiles are staged into LDS by the
// Tensor Data Mover (tensor_load_to_lds + s_wait_tensorcnt), using D# LDS
// padding to reproduce the bank-padded tile layout the WMMA fragment loaders
// expect. Input order (setup_inputs dict order, params flattened leaf-by-leaf
// in declaration order, w then b per layer):
//   0 scalar [32,5]         1 food [32,2,64,64]   2 food_relation [32,64,150]
//   3 thorn_relation [32,64,32,12]  4 thorn_mask [32,32]
//   5 clone [32,64,17]      6 clone_relation [32,64,64,12]  7 clone_mask [32,64]
//   8..11  scalar_enc   12..21 food_conv   22..25 food_rel_enc
//   26..29 thorn_rel_enc 30..33 clone_enc  34..37 clone_rel_enc
//   38..43 gcn1 (thorn|clone|agg)  44..49 gcn2  50..53 agg_enc
// ---------------------------------------------------------------------------

typedef __attribute__((ext_vector_type(16))) _Float16 v16h;
typedef __attribute__((ext_vector_type(8)))  _Float16 half8;
typedef __attribute__((ext_vector_type(8)))  float    v8f;
typedef __attribute__((ext_vector_type(4)))  unsigned int u32x4;
typedef __attribute__((ext_vector_type(8)))  int          i32x8;
typedef __attribute__((ext_vector_type(4)))  int          i32x4;

#define LDA40 40   // 32 K-halves (64B) + 8 pad (16B) -> 16B-aligned frags
#define LDK40 40
#define LDH72 72

#if defined(__has_builtin)
#  if __has_builtin(__builtin_amdgcn_tensor_load_to_lds)
#    define HAVE_TDM 1
#  endif
#endif
#ifndef HAVE_TDM
#  define HAVE_TDM 0
#endif

// ---- Tensor Data Mover: 2D f16 tile -> padded LDS tile --------------------
// Loads `rows` x `tile_k` f16 elements (row stride `row_stride` elements)
// from gptr into LDS at lds_byte_off, inserting 16B of padding after every
// 64B of data (pad_interval=16 DW, pad_amount=4 DW) == the LDA40 layout.
__device__ __forceinline__ void tdm_load_tile_f16(
    unsigned lds_byte_off, const _Float16* gptr, int tile_k, int rows,
    int row_stride) {
#if HAVE_TDM
  unsigned long long ga = (unsigned long long)(uintptr_t)gptr;
  u32x4 g0;
  g0[0] = 1u;                                            // count=1, user desc
  g0[1] = lds_byte_off;                                  // lds_addr (bytes)
  g0[2] = (unsigned)(ga & 0xffffffffu);                  // global_addr[31:0]
  g0[3] = (unsigned)((ga >> 32) & 0x01ffffffu) | (2u << 30);  // [56:32]|type=2
  i32x8 g1;
  g1[0] = (1 << 16)        // data_size = 2 bytes
        | (1 << 20)        // pad_enable
        | (3 << 22)        // pad_interval code 3 -> 16 DWORDs (64B)
        | (3 << 25);       // pad_amount   code 3 -> 4 DWORDs (16B)
  g1[1] = (row_stride & 0xffff) << 16;   // tensor_dim0[15:0] (abar_addr=0)
  g1[2] = (row_stride >> 16) & 0xffff;   // tensor_dim0[31:16], tdim1 lo = 0
  g1[3] = (tile_k << 16) | 2;            // tile_dim0 | tensor_dim1 hi (128Ki)
  g1[4] = rows & 0xffff;                 // tile_dim1 (tile_dim2 = 0)
  g1[5] = row_stride;                    // tensor_dim0_stride[31:0]
  g1[6] = 0;
  g1[7] = 0;
  i32x4 gz = {0, 0, 0, 0};
#if defined(__clang_major__) && __clang_major__ >= 23
  i32x8 gz8 = {0, 0, 0, 0, 0, 0, 0, 0};
  __builtin_amdgcn_tensor_load_to_lds(g0, g1, gz, gz, gz8, 0);
#else
  __builtin_amdgcn_tensor_load_to_lds(g0, g1, gz, gz, 0);
#endif
  __builtin_amdgcn_s_wait_tensorcnt(0);
#else
  (void)lds_byte_off; (void)gptr; (void)tile_k; (void)rows; (void)row_stride;
#endif
}

// ---- WMMA fragment helpers (ISA 7.12.2 layouts) ---------------------------
__device__ __forceinline__ v16h load_a_frag(const _Float16* S, int ld, int m0, int lane) {
  int l = lane & 15, hi = lane >> 4;
  const half8* rp = (const half8*)(S + (size_t)(m0 + l) * ld);
  half8 x = rp[hi];
  half8 y = rp[2 + hi];
  v16h a;
#pragma unroll
  for (int i = 0; i < 8; ++i) { a[i] = x[i]; a[8 + i] = y[i]; }
  return a;
}

__device__ __forceinline__ v16h load_b_frag(const _Float16* S, int ldk, int n0, int lane) {
  int n = n0 + (lane & 15), kh = lane >> 4;
  const half8* cp = (const half8*)(S + (size_t)n * ldk);
  half8 x = cp[kh], y = cp[2 + kh];
  v16h b;
#pragma unroll
  for (int i = 0; i < 8; ++i) { b[i] = x[i]; b[8 + i] = y[i]; }
  return b;
}

__device__ __forceinline__ v8f wmma_f16(v16h a, v16h b, v8f c) {
  return __builtin_amdgcn_wmma_f32_16x16x32_f16(
      false, a, false, b, (short)0, c, false, false);
}

// ---------------------------------------------------------------------------
// Fused 2-layer MLP (relu both layers): rows M x [d0 -> d1 -> 128], d1 <= 64.
// ---------------------------------------------------------------------------
template <typename OutT>
__global__ __launch_bounds__(256) void k_mlp2(
    const float* __restrict__ A, int M, int d0, int d1,
    const float* __restrict__ w1, const float* __restrict__ b1,
    const float* __restrict__ w2, const float* __restrict__ b2,
    OutT* __restrict__ out) {
  __shared__ __align__(16) _Float16 As[128 * LDA40];
  __shared__ __align__(16) _Float16 Bs[128 * LDK40];
  __shared__ __align__(16) _Float16 Hs[128 * LDH72];
  int tid = threadIdx.x, lane = tid & 31, wave = tid >> 5;
  int wm = wave & 3, wn = wave >> 2;
  size_t m_base = (size_t)blockIdx.x * 128;

  v8f acc1[2][2] = {};
  for (int k0 = 0; k0 < d0; k0 += 32) {
    for (int i = tid; i < 128 * 32; i += 256) {
      int r = i >> 5, c = i & 31;
      size_t gr = m_base + r; int gc = k0 + c;
      float v = (gr < (size_t)M && gc < d0) ? A[gr * d0 + gc] : 0.f;
      As[r * LDA40 + c] = (_Float16)v;
    }
    for (int i = tid; i < 64 * 32; i += 256) {
      int n = i >> 5, r = i & 31; int gk = k0 + r;
      float v = (gk < d0 && n < d1) ? w1[(size_t)gk * d1 + n] : 0.f;
      Bs[n * LDK40 + r] = (_Float16)v;
    }
    __syncthreads();
#pragma unroll
    for (int ms = 0; ms < 2; ++ms) {
      v16h a = load_a_frag(As, LDA40, wm * 32 + ms * 16, lane);
#pragma unroll
      for (int ns = 0; ns < 2; ++ns) {
        v16h b = load_b_frag(Bs, LDK40, wn * 32 + ns * 16, lane);
        acc1[ms][ns] = wmma_f16(a, b, acc1[ms][ns]);
      }
    }
    __syncthreads();
  }
  {
    int l = lane & 15, hi = lane >> 4;
#pragma unroll
    for (int ms = 0; ms < 2; ++ms)
#pragma unroll
      for (int ns = 0; ns < 2; ++ns) {
        int n = wn * 32 + ns * 16 + l;
        float bv = (n < d1) ? b1[n] : 0.f;
#pragma unroll
        for (int r = 0; r < 8; ++r) {
          int ml = wm * 32 + ms * 16 + hi * 8 + r;
          Hs[ml * LDH72 + n] = (_Float16)fmaxf(acc1[ms][ns][r] + bv, 0.f);
        }
      }
  }
  __syncthreads();

  v8f acc2[2][4] = {};
#pragma unroll
  for (int k0 = 0; k0 < 64; k0 += 32) {
    for (int i = tid; i < 128 * 32; i += 256) {
      int n = i >> 5, r = i & 31; int gk = k0 + r;
      float v = (gk < d1) ? w2[(size_t)gk * 128 + n] : 0.f;
      Bs[n * LDK40 + r] = (_Float16)v;
    }
    __syncthreads();
#pragma unroll
    for (int ms = 0; ms < 2; ++ms) {
      v16h a = load_a_frag(Hs + k0, LDH72, wm * 32 + ms * 16, lane);
#pragma unroll
      for (int ns = 0; ns < 4; ++ns) {
        v16h b = load_b_frag(Bs, LDK40, wn * 64 + ns * 16, lane);
        acc2[ms][ns] = wmma_f16(a, b, acc2[ms][ns]);
      }
    }
    __syncthreads();
  }
  {
    int l = lane & 15, hi = lane >> 4;
#pragma unroll
    for (int ms = 0; ms < 2; ++ms)
#pragma unroll
      for (int ns = 0; ns < 4; ++ns) {
        int n = wn * 64 + ns * 16 + l;
        float bv = b2[n];
#pragma unroll
        for (int r = 0; r < 8; ++r) {
          size_t m = m_base + wm * 32 + ms * 16 + hi * 8 + r;
          if (m < (size_t)M) out[m * 128 + n] = (OutT)fmaxf(acc2[ms][ns][r] + bv, 0.f);
        }
      }
  }
}

// ---------------------------------------------------------------------------
// GCN thorn branch: rows=(c,t), concat[clone(128)|tr_enc(128)] -> 128,
// relu * thorn_mask[t], max over t fused. Block = 4c x 32t = 128 rows, K=256.
// tr_enc K-steps (k0>=128) staged by the Tensor Data Mover.
// ---------------------------------------------------------------------------
__global__ __launch_bounds__(256) void k_gcn_thorn(
    const float* __restrict__ cl, const _Float16* __restrict__ tr_enc,
    const float* __restrict__ W, const float* __restrict__ bias,
    const float* __restrict__ tmask, float* __restrict__ trmax) {
  __shared__ __align__(16) _Float16 As[128 * LDA40];
  __shared__ __align__(16) _Float16 Bs[128 * LDK40];
  __shared__ float tm[32];
  int tid = threadIdx.x, lane = tid & 31, wave = tid >> 5;
  int wm = wave & 3, wn = wave >> 2;
  int b = blockIdx.x >> 4, c_base = (blockIdx.x & 15) * 4;
  if (tid < 32) tm[tid] = tmask[b * 32 + tid];
  unsigned as_off = (unsigned)(uintptr_t)(void*)As;
  const _Float16* tr_rows = tr_enc + ((size_t)b * 2048 + c_base * 32) * 128;

  v8f acc[2][4] = {};
  for (int k0 = 0; k0 < 256; k0 += 32) {
#if HAVE_TDM
    if (k0 >= 128) {
      if (wave == 0)        // TDM fills the whole 128x32 padded A tile
        tdm_load_tile_f16(as_off, tr_rows + (k0 - 128), 32, 128, 128);
    } else
#endif
    {
      for (int i = tid; i < 128 * 32; i += 256) {
        int r = i >> 5, c = i & 31;
        int cc = c_base + (r >> 5), t = r & 31, kf = k0 + c;
        _Float16 v;
        if (kf < 128) v = (_Float16)cl[((size_t)b * 64 + cc) * 128 + kf];
        else          v = tr_rows[(size_t)(r) * 128 + (kf - 128)];
        As[r * LDA40 + c] = v;
      }
    }
    for (int i = tid; i < 128 * 32; i += 256) {
      int n = i >> 5, r = i & 31;
      Bs[n * LDK40 + r] = (_Float16)W[(size_t)(k0 + r) * 128 + n];
    }
    __syncthreads();
#pragma unroll
    for (int ms = 0; ms < 2; ++ms) {
      v16h a = load_a_frag(As, LDA40, wm * 32 + ms * 16, lane);
#pragma unroll
      for (int ns = 0; ns < 4; ++ns) {
        v16h bfr = load_b_frag(Bs, LDK40, wn * 64 + ns * 16, lane);
        acc[ms][ns] = wmma_f16(a, bfr, acc[ms][ns]);
      }
    }
    __syncthreads();
  }
  int l = lane & 15, hi = lane >> 4;
  int c = c_base + wm;
#pragma unroll
  for (int ns = 0; ns < 4; ++ns) {
    int n = wn * 64 + ns * 16 + l;
    float bv = bias[n], pm = -1e30f;
#pragma unroll
    for (int ms = 0; ms < 2; ++ms)
#pragma unroll
      for (int r = 0; r < 8; ++r) {
        int t = ms * 16 + hi * 8 + r;
        pm = fmaxf(pm, fmaxf(acc[ms][ns][r] + bv, 0.f) * tm[t]);
      }
    pm = fmaxf(pm, __shfl_xor(pm, 16));
    if (lane < 16) trmax[((size_t)b * 64 + c) * 128 + n] = pm;
  }
}

// ---------------------------------------------------------------------------
// GCN clone branch: rows=(c1,c2), concat[cl(c1)|cl(c2)|cr_enc] (K=384) -> 128,
// relu * clone_mask[c2], max over c2. Block = 2 c1 x 64 c2 = 128 rows.
// cr_enc K-steps (k0>=256) staged by the Tensor Data Mover.
// ---------------------------------------------------------------------------
__global__ __launch_bounds__(256) void k_gcn_clone(
    const float* __restrict__ cl, const _Float16* __restrict__ cr_enc,
    const float* __restrict__ W, const float* __restrict__ bias,
    const float* __restrict__ cmask, float* __restrict__ crmax) {
  __shared__ __align__(16) _Float16 As[128 * LDA40];
  __shared__ __align__(16) _Float16 Bs[128 * LDK40];
  __shared__ float red[8 * 64];
  __shared__ float cm[64];
  int tid = threadIdx.x, lane = tid & 31, wave = tid >> 5;
  int wm = wave & 3, wn = wave >> 2;
  int b = blockIdx.x >> 5, c1g = blockIdx.x & 31;
  if (tid < 64) cm[tid] = cmask[b * 64 + tid];
  unsigned as_off = (unsigned)(uintptr_t)(void*)As;
  const _Float16* cr_rows = cr_enc + ((size_t)b * 4096 + c1g * 128) * 128;

  v8f acc[2][4] = {};
  for (int k0 = 0; k0 < 384; k0 += 32) {
#if HAVE_TDM
    if (k0 >= 256) {
      if (wave == 0)
        tdm_load_tile_f16(as_off, cr_rows + (k0 - 256), 32, 128, 128);
    } else
#endif
    {
      for (int i = tid; i < 128 * 32; i += 256) {
        int r = i >> 5, c = i & 31;
        int c1 = c1g * 2 + (r >> 6), c2 = r & 63, kf = k0 + c;
        _Float16 v;
        if (kf < 128)      v = (_Float16)cl[((size_t)b * 64 + c1) * 128 + kf];
        else if (kf < 256) v = (_Float16)cl[((size_t)b * 64 + c2) * 128 + (kf - 128)];
        else               v = cr_rows[(size_t)r * 128 + (kf - 256)];
        As[r * LDA40 + c] = v;
      }
    }
    for (int i = tid; i < 128 * 32; i += 256) {
      int n = i >> 5, r = i & 31;
      Bs[n * LDK40 + r] = (_Float16)W[(size_t)(k0 + r) * 128 + n];
    }
    __syncthreads();
#pragma unroll
    for (int ms = 0; ms < 2; ++ms) {
      v16h a = load_a_frag(As, LDA40, wm * 32 + ms * 16, lane);
#pragma unroll
      for (int ns = 0; ns < 4; ++ns) {
        v16h bfr = load_b_frag(Bs, LDK40, wn * 64 + ns * 16, lane);
        acc[ms][ns] = wmma_f16(a, bfr, acc[ms][ns]);
      }
    }
    __syncthreads();
  }
  int l = lane & 15, hi = lane >> 4;
#pragma unroll
  for (int ns = 0; ns < 4; ++ns) {
    int n = wn * 64 + ns * 16 + l;
    float bv = bias[n], pm = -1e30f;
#pragma unroll
    for (int ms = 0; ms < 2; ++ms)
#pragma unroll
      for (int r = 0; r < 8; ++r) {
        int c2 = (wm & 1) * 32 + ms * 16 + hi * 8 + r;
        pm = fmaxf(pm, fmaxf(acc[ms][ns][r] + bv, 0.f) * cm[c2]);
      }
    pm = fmaxf(pm, __shfl_xor(pm, 16));
    if (lane < 16) red[wave * 64 + ns * 16 + l] = pm;
  }
  __syncthreads();
  for (int i = tid; i < 2 * 128; i += 256) {
    int c1l = i >> 7, n = i & 127;
    int wnn = n >> 6, col = n & 63;
    float v = fmaxf(red[(c1l * 2 + 4 * wnn) * 64 + col],
                    red[(c1l * 2 + 1 + 4 * wnn) * 64 + col]);
    crmax[((size_t)b * 64 + c1g * 2 + c1l) * 128 + n] = v;
  }
}

// ---------------------------------------------------------------------------
// GCN aggregate: rows=b*c (2048), concat[cl|fr|trmax|crmax] (K=512) -> 128,
// out = cl + relu(gemm + bias). grid = 16.
// ---------------------------------------------------------------------------
__global__ __launch_bounds__(256) void k_gcn_agg(
    const float* __restrict__ cl, const float* __restrict__ fr,
    const float* __restrict__ trm, const float* __restrict__ crm,
    const float* __restrict__ W, const float* __restrict__ bias,
    float* __restrict__ out) {
  __shared__ __align__(16) _Float16 As[128 * LDA40];
  __shared__ __align__(16) _Float16 Bs[128 * LDK40];
  int tid = threadIdx.x, lane = tid & 31, wave = tid >> 5;
  int wm = wave & 3, wn = wave >> 2;
  size_t m_base = (size_t)blockIdx.x * 128;

  v8f acc[2][4] = {};
  for (int k0 = 0; k0 < 512; k0 += 32) {
    for (int i = tid; i < 128 * 32; i += 256) {
      int r = i >> 5, c = i & 31;
      size_t m = m_base + r; int kf = k0 + c;
      float v;
      if (kf < 128)      v = cl[m * 128 + kf];
      else if (kf < 256) v = fr[m * 128 + (kf - 128)];
      else if (kf < 384) v = trm[m * 128 + (kf - 256)];
      else               v = crm[m * 128 + (kf - 384)];
      As[r * LDA40 + c] = (_Float16)v;
    }
    for (int i = tid; i < 128 * 32; i += 256) {
      int n = i >> 5, r = i & 31;
      Bs[n * LDK40 + r] = (_Float16)W[(size_t)(k0 + r) * 128 + n];
    }
    if (k0 + 32 < 512) __builtin_prefetch(&W[(size_t)(k0 + 32) * 128 + tid]);
    __syncthreads();
#pragma unroll
    for (int ms = 0; ms < 2; ++ms) {
      v16h a = load_a_frag(As, LDA40, wm * 32 + ms * 16, lane);
#pragma unroll
      for (int ns = 0; ns < 4; ++ns) {
        v16h bfr = load_b_frag(Bs, LDK40, wn * 64 + ns * 16, lane);
        acc[ms][ns] = wmma_f16(a, bfr, acc[ms][ns]);
      }
    }
    __syncthreads();
  }
  int l = lane & 15, hi = lane >> 4;
#pragma unroll
  for (int ms = 0; ms < 2; ++ms)
#pragma unroll
    for (int ns = 0; ns < 4; ++ns) {
      int n = wn * 64 + ns * 16 + l;
      float bv = bias[n];
#pragma unroll
      for (int r = 0; r < 8; ++r) {
        size_t m = m_base + wm * 32 + ms * 16 + hi * 8 + r;
        out[m * 128 + n] = cl[m * 128 + n] + fmaxf(acc[ms][ns][r] + bv, 0.f);
      }
    }
}

// ---------------------------------------------------------------------------
// Generic single linear + relu, N = 128 (final agg_enc layers).
// ---------------------------------------------------------------------------
__global__ __launch_bounds__(256) void k_gemm_relu(
    const float* __restrict__ A, const float* __restrict__ W,
    const float* __restrict__ bias, float* __restrict__ C, int M, int K) {
  __shared__ __align__(16) _Float16 As[128 * LDA40];
  __shared__ __align__(16) _Float16 Bs[128 * LDK40];
  int tid = threadIdx.x, lane = tid & 31, wave = tid >> 5;
  int wm = wave & 3, wn = wave >> 2;
  size_t m_base = (size_t)blockIdx.x * 128;

  v8f acc[2][4] = {};
  for (int k0 = 0; k0 < K; k0 += 32) {
    for (int i = tid; i < 128 * 32; i += 256) {
      int r = i >> 5, c = i & 31;
      size_t gr = m_base + r; int gc = k0 + c;
      float v = (gr < (size_t)M && gc < K) ? A[gr * K + gc] : 0.f;
      As[r * LDA40 + c] = (_Float16)v;
    }
    for (int i = tid; i < 128 * 32; i += 256) {
      int n = i >> 5, r = i & 31; int gk = k0 + r;
      float v = (gk < K) ? W[(size_t)gk * 128 + n] : 0.f;
      Bs[n * LDK40 + r] = (_Float16)v;
    }
    __syncthreads();
#pragma unroll
    for (int ms = 0; ms < 2; ++ms) {
      v16h a = load_a_frag(As, LDA40, wm * 32 + ms * 16, lane);
#pragma unroll
      for (int ns = 0; ns < 4; ++ns) {
        v16h bfr = load_b_frag(Bs, LDK40, wn * 64 + ns * 16, lane);
        acc[ms][ns] = wmma_f16(a, bfr, acc[ms][ns]);
      }
    }
    __syncthreads();
  }
  int l = lane & 15, hi = lane >> 4;
#pragma unroll
  for (int ms = 0; ms < 2; ++ms)
#pragma unroll
    for (int ns = 0; ns < 4; ++ns) {
      int n = wn * 64 + ns * 16 + l;
      float bv = bias[n];
#pragma unroll
      for (int r = 0; r < 8; ++r) {
        size_t m = m_base + wm * 32 + ms * 16 + hi * 8 + r;
        if (m < (size_t)M) C[m * 128 + n] = fmaxf(acc[ms][ns][r] + bv, 0.f);
      }
    }
}

// ---------------------------------------------------------------------------
// Naive conv + relu (food path is ~0.2% of FLOPs; not worth WMMA im2col).
// ---------------------------------------------------------------------------
__global__ void k_conv_relu(const float* __restrict__ in, const float* __restrict__ w,
                            const float* __restrict__ bias, float* __restrict__ out,
                            int Ci, int Co, int Hi, int Wi, int Ho, int Wo,
                            int Kk, int S, int P, int total) {
  int idx = blockIdx.x * blockDim.x + threadIdx.x;
  if (idx >= total) return;
  int x = idx % Wo, y = (idx / Wo) % Ho, co = (idx / (Wo * Ho)) % Co,
      bb = idx / (Wo * Ho * Co);
  float s = bias[co];
  for (int ci = 0; ci < Ci; ++ci)
    for (int ky = 0; ky < Kk; ++ky) {
      int iy = y * S - P + ky;
      if ((unsigned)iy >= (unsigned)Hi) continue;
      for (int kx = 0; kx < Kk; ++kx) {
        int ix = x * S - P + kx;
        if ((unsigned)ix >= (unsigned)Wi) continue;
        s += in[(((size_t)bb * Ci + ci) * Hi + iy) * Wi + ix] *
             w[(((size_t)co * Ci + ci) * Kk + ky) * Kk + kx];
      }
    }
  out[idx] = fmaxf(s, 0.f);
}

// ---------------------------------------------------------------------------
// Final concat: [s | max_spatial(conv5) | max_c(cl*mask)] -> [32,384]
// ---------------------------------------------------------------------------
__global__ void k_finalcat(const float* __restrict__ s, const float* __restrict__ c5,
                           const float* __restrict__ clf, const float* __restrict__ cmask,
                           float* __restrict__ cat) {
  int b = blockIdx.x, n = threadIdx.x;   // 128 threads
  cat[(size_t)b * 384 + n] = s[(size_t)b * 128 + n];
  const float* cp = c5 + ((size_t)b * 128 + n) * 4;
  cat[(size_t)b * 384 + 128 + n] = fmaxf(fmaxf(cp[0], cp[1]), fmaxf(cp[2], cp[3]));
  float mx = -1e30f;
  for (int c = 0; c < 64; ++c)
    mx = fmaxf(mx, clf[((size_t)b * 64 + c) * 128 + n] * cmask[b * 64 + c]);
  cat[(size_t)b * 384 + 256 + n] = mx;
}

// ---------------------------------------------------------------------------
extern "C" void kernel_launch(void* const* d_in, const int* in_sizes, int n_in,
                              void* d_out, int out_size, void* d_ws, size_t ws_size,
                              hipStream_t stream) {
  (void)in_sizes; (void)n_in; (void)out_size; (void)ws_size;
  const float* scalar     = (const float*)d_in[0];
  const float* food       = (const float*)d_in[1];
  const float* food_rel   = (const float*)d_in[2];
  const float* thorn_rel  = (const float*)d_in[3];
  const float* thorn_mask = (const float*)d_in[4];
  const float* clone      = (const float*)d_in[5];
  const float* clone_rel  = (const float*)d_in[6];
  const float* clone_mask = (const float*)d_in[7];
  auto P = [&](int i) { return (const float*)d_in[i]; };

  // ---- workspace carving ----
  float* ws = (float*)d_ws; size_t off = 0;
  auto alloc = [&](size_t n) { float* p = ws + off; off += n; return p; };
  float* s_enc = alloc(32 * 128);
  float* c1 = alloc(32 * 32 * 16 * 16);
  float* c2 = alloc(32 * 64 * 8 * 8);
  float* c3 = alloc(32 * 64 * 4 * 4);
  float* c4 = alloc(32 * 128 * 2 * 2);
  float* c5 = alloc(32 * 128 * 2 * 2);
  float* fr    = alloc(2048 * 128);
  float* cl0   = alloc(2048 * 128);
  float* trmax = alloc(2048 * 128);
  float* crmax = alloc(2048 * 128);
  float* clA   = alloc(2048 * 128);
  float* clB   = alloc(2048 * 128);
  float* cat   = alloc(32 * 384);
  float* tmp   = alloc(32 * 128);
  _Float16* tr_enc = (_Float16*)(ws + off); off += (size_t)65536 * 128 / 2;
  _Float16* cr_enc = (_Float16*)(ws + off); off += (size_t)131072 * 128 / 2;

  dim3 blk(256);

  // ---- encoders (fused 2-layer MLPs on WMMA) ----
  k_mlp2<float><<<dim3(1), blk, 0, stream>>>(scalar, 32, 5, 32, P(8), P(9), P(10), P(11), s_enc);
  k_mlp2<float><<<dim3(16), blk, 0, stream>>>(food_rel, 2048, 150, 64, P(22), P(23), P(24), P(25), fr);
  k_mlp2<_Float16><<<dim3(512), blk, 0, stream>>>(thorn_rel, 65536, 12, 32, P(26), P(27), P(28), P(29), tr_enc);
  k_mlp2<float><<<dim3(16), blk, 0, stream>>>(clone, 2048, 17, 32, P(30), P(31), P(32), P(33), cl0);
  k_mlp2<_Float16><<<dim3(1024), blk, 0, stream>>>(clone_rel, 131072, 12, 32, P(34), P(35), P(36), P(37), cr_enc);

  // ---- food conv stack ----
  struct CD { int ci, co, hi, ho, k, s, p; };
  const CD cd[5] = {{2,32,64,16,5,4,2},{32,64,16,8,3,2,1},{64,64,8,4,3,2,1},
                    {64,128,4,2,3,2,1},{128,128,2,2,1,1,0}};
  const float* cin = food;
  float* couts[5] = {c1, c2, c3, c4, c5};
  for (int i = 0; i < 5; ++i) {
    int total = 32 * cd[i].co * cd[i].ho * cd[i].ho;
    k_conv_relu<<<dim3((total + 255) / 256), blk, 0, stream>>>(
        cin, P(12 + 2 * i), P(13 + 2 * i), couts[i],
        cd[i].ci, cd[i].co, cd[i].hi, cd[i].hi, cd[i].ho, cd[i].ho,
        cd[i].k, cd[i].s, cd[i].p, total);
    cin = couts[i];
  }

  // ---- GCN layer 1 ----
  k_gcn_thorn<<<dim3(512),  blk, 0, stream>>>(cl0, tr_enc, P(38), P(39), thorn_mask, trmax);
  k_gcn_clone<<<dim3(1024), blk, 0, stream>>>(cl0, cr_enc, P(40), P(41), clone_mask, crmax);
  k_gcn_agg  <<<dim3(16),   blk, 0, stream>>>(cl0, fr, trmax, crmax, P(42), P(43), clA);

  // ---- GCN layer 2 ----
  k_gcn_thorn<<<dim3(512),  blk, 0, stream>>>(clA, tr_enc, P(44), P(45), thorn_mask, trmax);
  k_gcn_clone<<<dim3(1024), blk, 0, stream>>>(clA, cr_enc, P(46), P(47), clone_mask, crmax);
  k_gcn_agg  <<<dim3(16),   blk, 0, stream>>>(clA, fr, trmax, crmax, P(48), P(49), clB);

  // ---- head ----
  k_finalcat<<<dim3(32), dim3(128), 0, stream>>>(s_enc, c5, clB, clone_mask, cat);
  k_gemm_relu<<<dim3(1), blk, 0, stream>>>(cat, P(50), P(51), tmp, 32, 384);
  k_gemm_relu<<<dim3(1), blk, 0, stream>>>(tmp, P(52), P(53), (float*)d_out, 32, 128);
}
